// DCompGCNCovLayer_15204184228278
// MI455X (gfx1250) — compile-verified, matrix operand order checked by reading, not run on hardware
//
#include <hip/hip_runtime.h>
#include <hip/hip_bf16.h>
#include <cmath>

// Problem constants (from reference)
#define V_N   50000
#define E_N   640000
#define IN_F  128
#define OUT_F 128
#define K_F   4
#define D_F   32
#define OD_F  32
#define NT_E  (E_N / 16)   // 40000 edge tiles of 16
#define NT_HALF (NT_E / 2) // first half -> in_w, second -> out_w
#define NT_V  (V_N / 16)   // 3125 node tiles

typedef __attribute__((ext_vector_type(16))) __bf16 v16bf;
typedef __attribute__((ext_vector_type(8)))  float  v8f;

union Frag16 { uint4 q[2]; v16bf v; };

// ---------------------------------------------------------------------------
// Kernel 0: zero h_sum (V*128) + stats (256) which are contiguous in ws
// ---------------------------------------------------------------------------
__global__ void zero_kernel(float* __restrict__ p, size_t n) {
  size_t i = (size_t)blockIdx.x * blockDim.x + threadIdx.x;
  size_t stride = (size_t)gridDim.x * blockDim.x;
  for (; i < n; i += stride) p[i] = 0.f;
}

// ---------------------------------------------------------------------------
// Kernel 1: precompute folded weights
//   Wc{In,Out}[n=k*32+od][i] = sum_d node_rel_w[k,i,d] * {in,out}_w[k,d,od]   (bf16, B^T layout)
//   v[k][i] = sum_d node_rel_w[k,i,d]*att_w[d]
//   u[k][i] = sum_d node_w[k,i,d]*att_w[32+d]
// Launch: 2 blocks x 256 (tid 0..511)
// ---------------------------------------------------------------------------
__global__ void precompute_kernel(const float* __restrict__ nrw,
                                  const float* __restrict__ nw,
                                  const float* __restrict__ in_w,
                                  const float* __restrict__ out_w,
                                  const float* __restrict__ att_w,
                                  __bf16* __restrict__ WcIn,
                                  __bf16* __restrict__ WcOut,
                                  float* __restrict__ uv) {
  int tid = blockIdx.x * blockDim.x + threadIdx.x;  // 0..511
  if (tid < 256) {
    int vr = tid >> 7;        // 0 = in, 1 = out
    int n  = tid & 127;       // k*32 + od
    int k = n >> 5, od = n & 31;
    const float* w2 = vr ? out_w : in_w;
    __bf16* dstp = vr ? WcOut : WcIn;
    for (int i = 0; i < IN_F; ++i) {
      float s = 0.f;
      const float* a = nrw + ((size_t)k * IN_F + i) * D_F;
      const float* b = w2 + (size_t)k * D_F * OD_F + od;
      #pragma unroll 8
      for (int d = 0; d < D_F; ++d) s += a[d] * b[d * OD_F];
      dstp[n * IN_F + i] = (__bf16)s;
    }
  }
  {
    int k = (tid >> 7) & 3, i = tid & 127;
    float v = 0.f, u = 0.f;
    const float* a = nrw + ((size_t)k * IN_F + i) * D_F;
    const float* b = nw  + ((size_t)k * IN_F + i) * D_F;
    #pragma unroll 8
    for (int d = 0; d < D_F; ++d) {
      v += a[d] * att_w[d];
      u += b[d] * att_w[D_F + d];
    }
    uv[k * IN_F + i]       = v;
    uv[512 + k * IN_F + i] = u;
  }
}

// ---------------------------------------------------------------------------
// Kernel 2: fused edge kernel.
// Per 16-edge tile per wave:
//   gather src*rel (edge_comp) into bf16 A-fragments + attention dots,
//   softmax over K, one 16x128x128 bf16 WMMA GEMM against folded Wc,
//   scale by att*norm, atomic scatter-add into h_sum[dst].
// The wave grid is split in halves so each wave's weight-LDS base (in_w vs
// out_w half) is loop-invariant. launch_bounds(256,4): >=4 waves/SIMD.
// ---------------------------------------------------------------------------
__global__ __launch_bounds__(256, 4) void edge_kernel(
    const float* __restrict__ node_repr, const float* __restrict__ rel_repr,
    const int* __restrict__ src, const int* __restrict__ dst,
    const int* __restrict__ etype, const float* __restrict__ norm,
    const __bf16* __restrict__ gWcIn, const __bf16* __restrict__ gWcOut,
    const float* __restrict__ guv, float* __restrict__ h_sum) {
  __shared__ __bf16 sWc[2][IN_F * OUT_F];  // 64 KB
  __shared__ float sUV[8 * IN_F];          // 4 KB: v[4][128] then u[4][128]
  {
    const uint4* a = (const uint4*)gWcIn;
    const uint4* b = (const uint4*)gWcOut;
    uint4* da = (uint4*)&sWc[0][0];
    uint4* db = (uint4*)&sWc[1][0];
    for (int i = threadIdx.x; i < (IN_F * OUT_F * 2 / 16); i += blockDim.x) {
      da[i] = a[i];
      db[i] = b[i];
    }
    for (int i = threadIdx.x; i < 8 * IN_F; i += blockDim.x) sUV[i] = guv[i];
  }
  __syncthreads();

  const int lane = threadIdx.x & 31;
  const int r  = lane & 15;   // edge row within tile / output column within n-tile
  const int hh = lane >> 4;   // which K-half this lane carries
  const int waveId = blockIdx.x * (blockDim.x >> 5) + (threadIdx.x >> 5);
  const int nWaves = gridDim.x * (blockDim.x >> 5);
  // static split: first half of waves -> tiles [0, NT_HALF) (in_w),
  // second half -> tiles [NT_HALF, NT_E) (out_w). sW is loop-invariant.
  const int nHalfWaves = nWaves >> 1;
  const int half = (waveId >= nHalfWaves) ? 1 : 0;
  const int wInHalf = waveId - half * nHalfWaves;
  const int tileEnd = (half + 1) * NT_HALF;
  const __bf16* sW = &sWc[half][0];

  for (int tile = half * NT_HALF + wInHalf; tile < tileEnd; tile += nHalfWaves) {
    const int e = tile * 16 + r;
    const int sIdx = src[e];
    const int dIdx = dst[e];
    const int eT   = etype[e];
    const float nrm = norm[e];
    const float* prow = node_repr + (size_t)sIdx * IN_F;
    const float* rrow = rel_repr + (size_t)eT * IN_F;
    const float* drow = node_repr + (size_t)dIdx * IN_F;

    float att[4] = {0.f, 0.f, 0.f, 0.f};
    v16bf afrag[4];

    // Pass 1: src*rel composition -> bf16 A fragments + edge part of the
    // attention logit (edge_comp . v[k]).
    #pragma unroll
    for (int kk = 0; kk < 4; ++kk) {
      const int i0 = kk * 32 + hh * 8;   // A layout: lane half hh carries K = hh*8..hh*8+7
      const int i1 = i0 + 16;            //           and K = 16+hh*8 .. 16+hh*8+7
      float4 s0 = *(const float4*)(prow + i0);
      float4 s1 = *(const float4*)(prow + i0 + 4);
      float4 s2 = *(const float4*)(prow + i1);
      float4 s3 = *(const float4*)(prow + i1 + 4);
      float4 q0 = *(const float4*)(rrow + i0);
      float4 q1 = *(const float4*)(rrow + i0 + 4);
      float4 q2 = *(const float4*)(rrow + i1);
      float4 q3 = *(const float4*)(rrow + i1 + 4);
      float pe[16] = {s0.x * q0.x, s0.y * q0.y, s0.z * q0.z, s0.w * q0.w,
                      s1.x * q1.x, s1.y * q1.y, s1.z * q1.z, s1.w * q1.w,
                      s2.x * q2.x, s2.y * q2.y, s2.z * q2.z, s2.w * q2.w,
                      s3.x * q3.x, s3.y * q3.y, s3.z * q3.z, s3.w * q3.w};
      #pragma unroll
      for (int j = 0; j < 16; ++j) afrag[kk][j] = (__bf16)pe[j];
      #pragma unroll
      for (int k = 0; k < 4; ++k) {
        const float* vk = &sUV[k * IN_F];
        #pragma unroll
        for (int j = 0; j < 8; ++j)
          att[k] += pe[j] * vk[i0 + j] + pe[8 + j] * vk[i1 + j];
      }
    }
    // Pass 2: dst-feature part of the attention logit (node_repr[dst] . u[k]).
    #pragma unroll
    for (int kk = 0; kk < 4; ++kk) {
      const int i0 = kk * 32 + hh * 8;
      const int i1 = i0 + 16;
      float4 d0 = *(const float4*)(drow + i0);
      float4 d1 = *(const float4*)(drow + i0 + 4);
      float4 d2 = *(const float4*)(drow + i1);
      float4 d3 = *(const float4*)(drow + i1 + 4);
      float pd[16] = {d0.x, d0.y, d0.z, d0.w, d1.x, d1.y, d1.z, d1.w,
                      d2.x, d2.y, d2.z, d2.w, d3.x, d3.y, d3.z, d3.w};
      #pragma unroll
      for (int k = 0; k < 4; ++k) {
        const float* uk = &sUV[4 * IN_F + k * IN_F];
        #pragma unroll
        for (int j = 0; j < 8; ++j)
          att[k] += pd[j] * uk[i0 + j] + pd[8 + j] * uk[i1 + j];
      }
    }

    // combine lane-pair partials, relu, softmax over K, fold in edge norm
    float mx = -1e30f;
    #pragma unroll
    for (int k = 0; k < 4; ++k) {
      float x = att[k];
      x += __shfl_xor(x, 16, 32);
      x = fmaxf(x, 0.f);
      att[k] = x;
      mx = fmaxf(mx, x);
    }
    float ssum = 0.f;
    #pragma unroll
    for (int k = 0; k < 4; ++k) { att[k] = __expf(att[k] - mx); ssum += att[k]; }
    const float inv = nrm / ssum;
    #pragma unroll
    for (int k = 0; k < 4; ++k) att[k] *= inv;

    // precompute the 8 atomic base pointers once; the per-nt column step
    // (nt*16 floats) folds into the instruction's immediate offset.
    float* basep[8];
    #pragma unroll
    for (int i = 0; i < 8; ++i) {
      const int rd = __shfl(dIdx, (hh << 3) + i, 32);
      basep[i] = h_sum + ((size_t)rd << 7) + r;
    }

    // k-outer so only 8 broadcast att values are live at a time
    #pragma unroll
    for (int k = 0; k < 4; ++k) {
      float rowAttK[8];
      #pragma unroll
      for (int i = 0; i < 8; ++i)
        rowAttK[i] = __shfl(att[k], (hh << 3) + i, 32);
      #pragma unroll
      for (int t2 = 0; t2 < 2; ++t2) {
        const int nt = k * 2 + t2;
        v8f acc = {0.f, 0.f, 0.f, 0.f, 0.f, 0.f, 0.f, 0.f};
        #pragma unroll
        for (int kk = 0; kk < 4; ++kk) {
          const __bf16* bp = sW + ((nt * 16 + r) * IN_F + kk * 32 + hh * 8);
          Frag16 f;
          f.q[0] = *(const uint4*)(bp);
          f.q[1] = *(const uint4*)(bp + 16);
          acc = __builtin_amdgcn_wmma_f32_16x16x32_bf16(
              false, afrag[kk], false, f.v, (short)0, acc, false, false);
        }
        // scale first into distinct registers, then issue all 8 atomics
        // back-to-back (single clause, no per-atomic WAR waits)
        float vals[8];
        #pragma unroll
        for (int i = 0; i < 8; ++i) vals[i] = acc[i] * rowAttK[i];
        #pragma unroll
        for (int i = 0; i < 8; ++i) atomicAdd(basep[i] + nt * 16, vals[i]);
      }
    }
  }
}

// ---------------------------------------------------------------------------
// Kernel 3: self-loop (WMMA) + bias, in-place h -> h2, BN stats via atomics.
//   h2 = h/3 + ((h/3)*loop_rel) @ loop_w / 3 + bias
// ---------------------------------------------------------------------------
__global__ __launch_bounds__(256, 4) void selfloop_bn_kernel(
    float* __restrict__ h, const float* __restrict__ loop_rel,
    const float* __restrict__ loop_w, const float* __restrict__ bias,
    float* __restrict__ stats) {
  __shared__ __bf16 sW[IN_F * OUT_F];  // loop_w^T as bf16 B-fragments (32 KB)
  for (int idx = threadIdx.x; idx < IN_F * OUT_F; idx += blockDim.x) {
    int o = idx >> 7, i = idx & 127;
    sW[idx] = (__bf16)loop_w[i * OUT_F + o];
  }
  __syncthreads();

  const int lane = threadIdx.x & 31;
  const int r  = lane & 15;
  const int hh = lane >> 4;
  const int waveId = blockIdx.x * (blockDim.x >> 5) + (threadIdx.x >> 5);
  const int nWaves = gridDim.x * (blockDim.x >> 5);
  const float third = 1.f / 3.f;

  float sA[8] = {0.f, 0.f, 0.f, 0.f, 0.f, 0.f, 0.f, 0.f};
  float sQ[8] = {0.f, 0.f, 0.f, 0.f, 0.f, 0.f, 0.f, 0.f};

  for (int tile = waveId; tile < NT_V; tile += nWaves) {
    const int node = tile * 16 + r;
    const float* hrow = h + (size_t)node * OUT_F;
    v16bf afrag[4];
    #pragma unroll
    for (int kk = 0; kk < 4; ++kk) {
      const int i0 = kk * 32 + hh * 8;
      const int i1 = i0 + 16;
      float4 a0 = *(const float4*)(hrow + i0);
      float4 a1 = *(const float4*)(hrow + i0 + 4);
      float4 a2 = *(const float4*)(hrow + i1);
      float4 a3 = *(const float4*)(hrow + i1 + 4);
      float4 l0 = *(const float4*)(loop_rel + i0);
      float4 l1 = *(const float4*)(loop_rel + i0 + 4);
      float4 l2 = *(const float4*)(loop_rel + i1);
      float4 l3 = *(const float4*)(loop_rel + i1 + 4);
      float t[16] = {a0.x * l0.x * third, a0.y * l0.y * third, a0.z * l0.z * third, a0.w * l0.w * third,
                     a1.x * l1.x * third, a1.y * l1.y * third, a1.z * l1.z * third, a1.w * l1.w * third,
                     a2.x * l2.x * third, a2.y * l2.y * third, a2.z * l2.z * third, a2.w * l2.w * third,
                     a3.x * l3.x * third, a3.y * l3.y * third, a3.z * l3.z * third, a3.w * l3.w * third};
      #pragma unroll
      for (int j = 0; j < 16; ++j) afrag[kk][j] = (__bf16)t[j];
    }
    #pragma unroll
    for (int nt = 0; nt < 8; ++nt) {
      v8f acc = {0.f, 0.f, 0.f, 0.f, 0.f, 0.f, 0.f, 0.f};
      #pragma unroll
      for (int kk = 0; kk < 4; ++kk) {
        const __bf16* bp = sW + ((nt * 16 + r) * IN_F + kk * 32 + hh * 8);
        Frag16 f;
        f.q[0] = *(const uint4*)(bp);
        f.q[1] = *(const uint4*)(bp + 16);
        acc = __builtin_amdgcn_wmma_f32_16x16x32_bf16(
            false, afrag[kk], false, f.v, (short)0, acc, false, false);
      }
      const int col = nt * 16 + r;
      const float bcol = bias[col];
      #pragma unroll
      for (int i = 0; i < 8; ++i) {
        const int rowNode = tile * 16 + (hh << 3) + i;
        const size_t off = (size_t)rowNode * OUT_F + col;
        const float h2 = h[off] * third + acc[i] * third + bcol;
        h[off] = h2;  // in-place: this wave owns these rows exclusively
        sA[nt] += h2;
        sQ[nt] += h2 * h2;
      }
    }
  }
  // per-column stats: combine lane pair, then one atomic per column per wave
  #pragma unroll
  for (int nt = 0; nt < 8; ++nt) {
    float a = sA[nt] + __shfl_xor(sA[nt], 16, 32);
    float q = sQ[nt] + __shfl_xor(sQ[nt], 16, 32);
    if (hh == 0) {
      atomicAdd(&stats[nt * 16 + r], a);
      atomicAdd(&stats[128 + nt * 16 + r], q);
    }
  }
}

// ---------------------------------------------------------------------------
// Kernel 4: BatchNorm finalize + tanh
// ---------------------------------------------------------------------------
__global__ void finalize_kernel(const float* __restrict__ h,
                                const float* __restrict__ stats,
                                const float* __restrict__ gamma,
                                const float* __restrict__ beta,
                                float* __restrict__ out) {
  size_t i = (size_t)blockIdx.x * blockDim.x + threadIdx.x;
  if (i >= (size_t)V_N * OUT_F) return;
  const int c = (int)(i & 127);
  const float mean = stats[c] * (1.f / V_N);
  const float var = stats[128 + c] * (1.f / V_N) - mean * mean;
  const float x = (h[i] - mean) * rsqrtf(var + 1e-5f);
  out[i] = tanhf(gamma[c] * x + beta[c]);
}

// ---------------------------------------------------------------------------
// Kernel 5: rel_out = rel_repr @ w_rel   (474 x 128 x 128, tiny)
// ---------------------------------------------------------------------------
__global__ void rel_kernel(const float* __restrict__ rel,
                           const float* __restrict__ w_rel,
                           float* __restrict__ out) {
  int idx = blockIdx.x * blockDim.x + threadIdx.x;
  if (idx >= 474 * 128) return;
  int row = idx >> 7, c = idx & 127;
  float s = 0.f;
  #pragma unroll 8
  for (int i = 0; i < 128; ++i) s += rel[row * 128 + i] * w_rel[i * 128 + c];
  out[idx] = s;
}

// ---------------------------------------------------------------------------
extern "C" void kernel_launch(void* const* d_in, const int* in_sizes, int n_in,
                              void* d_out, int out_size, void* d_ws, size_t ws_size,
                              hipStream_t stream) {
  (void)in_sizes; (void)n_in; (void)out_size; (void)ws_size;
  const float* node_repr  = (const float*)d_in[0];
  const float* rel_repr   = (const float*)d_in[1];
  const int*   src        = (const int*)d_in[2];
  const int*   dst        = (const int*)d_in[3];
  const int*   etype      = (const int*)d_in[4];
  const float* norm       = (const float*)d_in[5];
  const float* node_w     = (const float*)d_in[6];
  const float* node_rel_w = (const float*)d_in[7];
  const float* in_w       = (const float*)d_in[8];
  const float* out_w      = (const float*)d_in[9];
  const float* att_w      = (const float*)d_in[10];
  const float* loop_rel   = (const float*)d_in[11];
  const float* loop_w     = (const float*)d_in[12];
  const float* w_rel      = (const float*)d_in[13];
  const float* bias       = (const float*)d_in[14];
  const float* bn_gamma   = (const float*)d_in[15];
  const float* bn_beta    = (const float*)d_in[16];

  // workspace layout (~25.7 MB)
  float* h_sum = (float*)d_ws;                        // V*128 f32
  float* stats = h_sum + (size_t)V_N * OUT_F;         // 256 f32
  float* uv    = stats + 256;                         // 1024 f32 (v then u)
  __bf16* WcIn  = (__bf16*)(uv + 1024);               // 128*128 bf16
  __bf16* WcOut = WcIn + IN_F * OUT_F;                // 128*128 bf16

  zero_kernel<<<512, 256, 0, stream>>>(h_sum, (size_t)V_N * OUT_F + 256);
  precompute_kernel<<<2, 256, 0, stream>>>(node_rel_w, node_w, in_w, out_w,
                                           att_w, WcIn, WcOut, uv);
  edge_kernel<<<512, 256, 0, stream>>>(node_repr, rel_repr, src, dst, etype,
                                       norm, WcIn, WcOut, uv, h_sum);
  selfloop_bn_kernel<<<128, 256, 0, stream>>>(h_sum, loop_rel, loop_w, bias, stats);
  finalize_kernel<<<(V_N * OUT_F) / 256, 256, 0, stream>>>(h_sum, stats, bn_gamma,
                                                           bn_beta, (float*)d_out);
  rel_kernel<<<(474 * 128 + 255) / 256, 256, 0, stream>>>(
      rel_repr, w_rel, (float*)d_out + (size_t)V_N * OUT_F);
}
